// RandomProjectionQuantizer_28243704938613
// MI455X (gfx1250) — matrix operand order repaired and checked by reading, not compiled, not run on hardware
//
#include <hip/hip_runtime.h>
#include <hip/hip_bf16.h>
#include <stdint.h>

// ---- CDNA5 (gfx1250) vector types ----
typedef __attribute__((ext_vector_type(16))) __bf16         v16bf;
typedef __attribute__((ext_vector_type(8)))  float          v8f;
typedef __attribute__((ext_vector_type(4)))  float          v4f;
typedef __attribute__((ext_vector_type(16))) unsigned short v16u;
typedef __attribute__((ext_vector_type(8)))  unsigned short v8u;
typedef __attribute__((ext_vector_type(4)))  unsigned short v4u;
typedef __attribute__((ext_vector_type(4)))  unsigned int   u32x4;
typedef __attribute__((ext_vector_type(8)))  unsigned int   u32x8;

#define INPUT_DIM       512
#define CB_DIM          16
#define CB_VOCAB        8192
#define ROWS_PER_WAVE   16
#define WAVES_PER_BLOCK 8
#define ROWS_PER_BLOCK  (ROWS_PER_WAVE * WAVES_PER_BLOCK)   // 128
#define BLOCK_THREADS   256
#define CB_USHORTS      (CB_VOCAB * CB_DIM)                 // 131072 (256 KB as bf16)
#define XP_USHORTS      (ROWS_PER_BLOCK * CB_DIM)           // 2048
#define PAD_USHORTS     16                                  // 32B zero pad for B upper-K half
#define LDS_USHORTS     (CB_USHORTS + XP_USHORTS + PAD_USHORTS)
#define TDM_ELEMS_8B    (CB_USHORTS * 2 / 8)                // 32768 8-byte elements

__device__ __forceinline__ unsigned short f32_to_bf16_rne(float f) {
    unsigned int u = __float_as_uint(f);
    u += 0x7FFFu + ((u >> 16) & 1u);           // round-to-nearest-even
    return (unsigned short)(u >> 16);
}

// ---------- Kernel 1: CB f32 -> bf16 bits in workspace (one-time, 0.5 MB read) ----------
__global__ __launch_bounds__(BLOCK_THREADS)
void cb_to_bf16_kernel(const float* __restrict__ CB, unsigned short* __restrict__ cbb) {
    const int i = (blockIdx.x * BLOCK_THREADS + threadIdx.x) * 4;
    v4f v = *(const v4f*)(CB + i);
    v4u p;
    p[0] = f32_to_bf16_rne(v[0]);
    p[1] = f32_to_bf16_rne(v[1]);
    p[2] = f32_to_bf16_rne(v[2]);
    p[3] = f32_to_bf16_rne(v[3]);
    *(v4u*)(cbb + i) = p;
}

// ---------- Kernel 2: fused project+normalize+argmax over codebook ----------
__global__ __launch_bounds__(BLOCK_THREADS, 1)
void rpq_wmma_kernel(const float* __restrict__ x,
                     const float* __restrict__ P,
                     const unsigned short* __restrict__ cbb,   // bf16 CB in ws
                     int* __restrict__ out) {
    extern __shared__ unsigned short smem[];                 // ~260 KB of 320 KB WGP LDS
    unsigned short* ldsCB  = smem;                           // [8192][16] bf16 bits
    unsigned short* ldsXP  = smem + CB_USHORTS;              // [128][16]  bf16 bits
    unsigned short* ldsPad = smem + CB_USHORTS + XP_USHORTS; // 32B of zeros

    const int tid      = threadIdx.x;
    const int lane     = tid & 31;
    const int wid      = tid >> 5;
    const int wave_row = blockIdx.x * ROWS_PER_BLOCK + wid * ROWS_PER_WAVE;

    // ---- TDM: one descriptor-driven DMA of the whole bf16 codebook into LDS ----
    // 2D tile of 32768 x 1 8-byte elements (256 KB contiguous). Issued by wave 0
    // only (TDM ignores EXEC, SGPR-sourced), tracked by TENSORcnt; overlaps phase 1.
    if (wid == 0) {
        const unsigned            lds_off = (unsigned)(size_t)ldsCB;  // wave-relative LDS byte addr
        const unsigned long long  ga      = (unsigned long long)(size_t)cbb;

        u32x4 g0;
        g0[0] = 1u;                                             // count=1, user D#, no gather
        g0[1] = lds_off;                                        // lds_addr [63:32]
        g0[2] = (unsigned)ga;                                   // global_addr[31:0]
        g0[3] = (unsigned)((ga >> 32) & 0x01FFFFFFull)          // global_addr[56:32]
              | 0x80000000u;                                    // type=2 ("image") [127:126]

        u32x8 g1;
        g1[0] = 3u << 16;                                       // wg_mask=0, data_size=3 (8B)
        g1[1] = ((unsigned)TDM_ELEMS_8B & 0xFFFFu) << 16;       // tensor_dim0[15:0] at [63:48]
        g1[2] = ((unsigned)TDM_ELEMS_8B >> 16)                  // tensor_dim0[31:16] at [79:64]
              | (1u << 16);                                     // tensor_dim1[15:0] = 1
        g1[3] = ((unsigned)TDM_ELEMS_8B & 0xFFFFu) << 16;       // tile_dim0 at [127:112]
        g1[4] = 1u;                                             // tile_dim1 = 1, tile_dim2 = 0
        g1[5] = (unsigned)TDM_ELEMS_8B;                         // tensor_dim0_stride[31:0]
        g1[6] = 0u;                                             // stride0[47:32], stride1[15:0]
        g1[7] = 0u;                                             // tensor_dim1_stride[47:16]

        asm volatile("tensor_load_to_lds %0, %1" :: "s"(g0), "s"(g1) : "memory");
    }
    if (tid < PAD_USHORTS) ldsPad[tid] = 0;                  // zero half for K=16..31

    // ---------------- Phase 1: xp = normalize(x @ P), f32 exact ----------------
    // One wave per row: lane owns K-slice [lane*16, lane*16+16), 16 f32 accumulators.
    for (int rr = 0; rr < ROWS_PER_WAVE; ++rr) {
        const float* xr = x + (size_t)(wave_row + rr) * INPUT_DIM;
        __builtin_prefetch(xr + INPUT_DIM, 0, 1);            // global_prefetch_b8

        float acc[CB_DIM];
#pragma unroll
        for (int c = 0; c < CB_DIM; ++c) acc[c] = 0.0f;

#pragma unroll
        for (int kk = 0; kk < 4; ++kk) {
            v4f xv = *(const v4f*)(xr + lane * 16 + kk * 4);
#pragma unroll
            for (int j = 0; j < 4; ++j) {
                const float xs = xv[j];
                const float* Pr = P + (lane * 16 + kk * 4 + j) * CB_DIM;
#pragma unroll
                for (int cc = 0; cc < 4; ++cc) {
                    v4f pv = *(const v4f*)(Pr + cc * 4);
#pragma unroll
                    for (int c2 = 0; c2 < 4; ++c2)
                        acc[cc * 4 + c2] = fmaf(xs, pv[c2], acc[cc * 4 + c2]);
                }
            }
        }
        // wave32 butterfly reduction of each of the 16 partial dots
#pragma unroll
        for (int c = 0; c < CB_DIM; ++c) {
#pragma unroll
            for (int off = 16; off >= 1; off >>= 1)
                acc[c] += __shfl_xor(acc[c], off, 32);
        }
        if (lane == 0) {
            float ss = 0.0f;
#pragma unroll
            for (int c = 0; c < CB_DIM; ++c) ss = fmaf(acc[c], acc[c], ss);
            const float inv = 1.0f / fmaxf(sqrtf(ss), 1e-12f);
            v16u pk;
#pragma unroll
            for (int c = 0; c < CB_DIM; ++c) pk[c] = f32_to_bf16_rne(acc[c] * inv);
            *(v16u*)(ldsXP + (wid * ROWS_PER_WAVE + rr) * CB_DIM) = pk;
        }
    }

    __builtin_amdgcn_s_wait_tensorcnt(0);   // TDM done (nonzero count only in wave 0)
    __syncthreads();                        // DMA + xp staging visible to all waves

    // ---------------- Phase 2: sim = xp @ CB^T via v_wmma_f32_16x16x32_bf16 ----------------
    // A (16x32 bf16): lane m=lane&15; lanes 0-15 carry K=0..15, lanes 16-31 (K=16..31)
    // zeroed -> K padded 16->32.
    const int m    = lane & 15;
    v16u au = {};
    if (lane < 16) {
        au = *(const v16u*)(ldsXP + (wid * ROWS_PER_WAVE + m) * CB_DIM);
    }
    const v16bf A = __builtin_bit_cast(v16bf, au);

    float best[8];
    int   bidx[8];
#pragma unroll
    for (int r = 0; r < 8; ++r) { best[r] = -3.402823466e38f; bidx[r] = 0; }

    // B (32x16 bf16): lane n<16 walks codebook rows (stride 16 rows); lanes 16-31
    // read the fixed 32B zero pad (K=16..31 half) -> unconditional ds_load_b128 x2.
    const unsigned short* bptr = (lane < 16) ? (ldsCB + lane * CB_DIM) : ldsPad;
    const int bstep = (lane < 16) ? (16 * CB_DIM) : 0;
    const int mycol = lane & 15;

    for (int cb = 0; cb < CB_VOCAB; cb += 16) {
        const v16u bu = *(const v16u*)bptr;
        bptr += bstep;
        const v16bf B = __builtin_bit_cast(v16bf, bu);

        v8f c0 = {};
        v8f d = __builtin_amdgcn_wmma_f32_16x16x32_bf16(
            /*neg_a=*/false, A, /*neg_b=*/false, B,
            /*c_mod=*/(short)0, c0, /*reuse_a=*/false, /*reuse_b=*/false);

        // D layout: vgpr r, lanes 0-15 -> (row r, col lane); lanes 16-31 -> (row r+8, col lane-16)
        const int col = cb + mycol;
#pragma unroll
        for (int r = 0; r < 8; ++r) {
            const bool p = d[r] > best[r];
            best[r] = p ? d[r] : best[r];
            bidx[r] = p ? col  : bidx[r];
        }
    }

    // argmax across the 16 lanes of each half (ties -> lowest index, like jnp.argmax)
#pragma unroll
    for (int r = 0; r < 8; ++r) {
        float v = best[r];
        int   i = bidx[r];
#pragma unroll
        for (int off = 8; off >= 1; off >>= 1) {
            const float ov = __shfl_xor(v, off, 32);
            const int   oi = __shfl_xor(i, off, 32);
            if (ov > v || (ov == v && oi < i)) { v = ov; i = oi; }
        }
        bidx[r] = i;
    }

    if (lane == 0 || lane == 16) {
        const int base = wave_row + ((lane >> 4) << 3);
#pragma unroll
        for (int r = 0; r < 8; ++r) out[base + r] = bidx[r];
    }
}

extern "C" void kernel_launch(void* const* d_in, const int* in_sizes, int n_in,
                              void* d_out, int out_size, void* d_ws, size_t ws_size,
                              hipStream_t stream) {
    (void)n_in; (void)ws_size; (void)out_size;
    const float* x  = (const float*)d_in[0];   // [N, 512]
    const float* P  = (const float*)d_in[1];   // [512, 16]
    const float* CB = (const float*)d_in[2];   // [8192, 16]
    int* out = (int*)d_out;                    // [N] int32 indices
    unsigned short* cbb = (unsigned short*)d_ws;   // [8192*16] bf16 bits (256 KB)

    const int N      = in_sizes[0] / INPUT_DIM;    // 16384
    const int blocks = N / ROWS_PER_BLOCK;         // 128

    // Kernel 1: one-time CB f32 -> bf16 (131072 elems, 4/thread)
    cb_to_bf16_kernel<<<(CB_USHORTS / 4 + BLOCK_THREADS - 1) / BLOCK_THREADS,
                        BLOCK_THREADS, 0, stream>>>(CB, cbb);

    // Kernel 2: TDM-staged codebook + fused projection + WMMA similarity + argmax
    const size_t lds_bytes = (size_t)LDS_USHORTS * sizeof(unsigned short); // ~260 KB
    rpq_wmma_kernel<<<blocks, BLOCK_THREADS, lds_bytes, stream>>>(x, P, cbb, out);
}
